// Argus_67748814127519
// MI455X (gfx1250) — matrix operand order repaired and checked by reading, not compiled, not run on hardware
//
#include <hip/hip_runtime.h>
#include <math.h>

typedef __attribute__((ext_vector_type(16))) _Float16 v16h;
typedef __attribute__((ext_vector_type(8)))  float    v8f;

#define NN 50000     // nodes
#define NE 200000    // edges
#define SS 5         // samples

// ---------------- Threefry-2x32-20 (JAX-compatible) ----------------
__host__ __device__ inline unsigned rotl32(unsigned x, int d) {
  return (x << d) | (x >> (32 - d));
}
__host__ __device__ inline void threefry2x32(unsigned k0, unsigned k1,
                                             unsigned x0, unsigned x1,
                                             unsigned* o0, unsigned* o1) {
  const unsigned ks2 = k0 ^ k1 ^ 0x1BD11BDAu;
  x0 += k0; x1 += k1;
#define TF_R4(a,b,c,d) \
  x0 += x1; x1 = rotl32(x1, a); x1 ^= x0; \
  x0 += x1; x1 = rotl32(x1, b); x1 ^= x0; \
  x0 += x1; x1 = rotl32(x1, c); x1 ^= x0; \
  x0 += x1; x1 = rotl32(x1, d); x1 ^= x0;
  TF_R4(13,15,26,6)  x0 += k1;  x1 += ks2 + 1u;
  TF_R4(17,29,16,24) x0 += ks2; x1 += k0 + 2u;
  TF_R4(13,15,26,6)  x0 += k0;  x1 += k1 + 3u;
  TF_R4(17,29,16,24) x0 += k1;  x1 += ks2 + 4u;
  TF_R4(13,15,26,6)  x0 += ks2; x1 += k0 + 5u;
#undef TF_R4
  *o0 = x0; *o1 = x1;
}

// ---------------- tiny elementwise kernels ----------------
__global__ void Argus_fill(float* p, float v, int n) {
  int i = blockIdx.x * blockDim.x + threadIdx.x;
  if (i < n) p[i] = v;
}
__global__ void Argus_relu(float* p, int n) {
  int i = blockIdx.x * blockDim.x + threadIdx.x;
  if (i < n) p[i] = fmaxf(p[i], 0.0f);
}
__global__ void Argus_count_deg(const int* __restrict__ ei, float* __restrict__ deg) {
  int e = blockIdx.x * blockDim.x + threadIdx.x;
  if (e < NE) atomicAdd(&deg[ei[NE + e]], 1.0f);
}
__global__ void Argus_dinv(const float* __restrict__ deg, float* __restrict__ dinv) {
  int i = blockIdx.x * blockDim.x + threadIdx.x;
  if (i < NN) dinv[i] = rsqrtf(deg[i]);   // deg >= 1 (self loop)
}

// ---------------- B-fragment packing (f32 -> f16, WMMA B layout) ----------------
// B (32x16 per WMMA): lane l holds column n=l&15; half i -> K = (i<8 ? 8h+i : 16+8h+(i-8)), h = l>=16
__global__ void Argus_packB(const float* __restrict__ W, _Float16* __restrict__ frag, int ksteps) {
  int t = blockIdx.x * blockDim.x + threadIdx.x;      // t = (s*2+c)*32 + l
  if (t >= ksteps * 64) return;
  int l = t & 31, sc = t >> 5, c = sc & 1, s = sc >> 1;
  int ncol = c * 16 + (l & 15);
  for (int i = 0; i < 16; ++i) {
    int K = s * 32 + ((l < 16) ? 0 : 8) + ((i < 8) ? i : i + 8);
    frag[(size_t)t * 16 + i] = (_Float16)W[(size_t)K * 32 + ncol];
  }
}
// Packed [nnW2 ; nnb2] as virtual 288x32 matrix (K<256: nnW2[K/32][ (K%32)*32+col ])
__global__ void Argus_packNN(const float* __restrict__ nnW2, const float* __restrict__ nnb2,
                             _Float16* __restrict__ frag) {
  int t = blockIdx.x * blockDim.x + threadIdx.x;
  if (t >= 9 * 64) return;
  int l = t & 31, sc = t >> 5, c = sc & 1, s = sc >> 1;
  int ncol = c * 16 + (l & 15);
  for (int i = 0; i < 16; ++i) {
    int K = s * 32 + ((l < 16) ? 0 : 8) + ((i < 8) ? i : i + 8);
    float v = (K < 256) ? nnW2[(size_t)(K >> 5) * 1024 + (size_t)(K & 31) * 32 + ncol]
                        : nnb2[(size_t)(K - 256) * 32 + ncol];
    frag[(size_t)t * 16 + i] = (_Float16)v;
  }
}

// ---------------- WMMA GEMM: C(nrows x 32) = A(nrows x 32*ksteps) @ Bfrag ----------------
__global__ void Argus_gemm16(const float* __restrict__ A, const _Float16* __restrict__ Bfrag,
                             float* __restrict__ C, int nrows, int ksteps) {
  int lane = threadIdx.x & 31;
  int wv   = threadIdx.x >> 5;
  int tile = blockIdx.x * 8 + wv;
  if (tile * 16 >= nrows) return;
  int row0  = tile * 16;
  int m     = lane & 15;
  int kbase = (lane < 16) ? 0 : 8;      // A layout: lane-half selects K sub-range
  const v16h* Bf = (const v16h*)Bfrag;
  const int K = ksteps * 32;
  const float* Arow = A + (size_t)(row0 + m) * K;
  v8f acc0 = {}, acc1 = {};
  for (int s = 0; s < ksteps; ++s) {
    v16h a;
#pragma unroll
    for (int i = 0; i < 16; ++i) {
      int k = s * 32 + kbase + ((i < 8) ? i : i + 8);
      a[i] = (_Float16)Arow[k];
    }
    v16h b0 = Bf[(s * 2 + 0) * 32 + lane];
    v16h b1 = Bf[(s * 2 + 1) * 32 + lane];
    acc0 = __builtin_amdgcn_wmma_f32_16x16x32_f16(false, a, false, b0, (short)0, acc0, false, false);
    acc1 = __builtin_amdgcn_wmma_f32_16x16x32_f16(false, a, false, b1, (short)0, acc1, false, false);
  }
  int n = lane & 15;
  int mbase = (lane < 16) ? 0 : 8;       // C/D layout
#pragma unroll
  for (int v = 0; v < 8; ++v) {
    int r = row0 + mbase + v;
    C[(size_t)r * 32 + n]      = acc0[v];
    C[(size_t)r * 32 + 16 + n] = acc1[v];
  }
}

// ---------------- GCN: self-loop init + edge scatter ----------------
__global__ void Argus_gcn_init(float* __restrict__ out, const float* __restrict__ h,
                               const float* __restrict__ dinv, const float* __restrict__ bias) {
  int i = blockIdx.x * blockDim.x + threadIdx.x;
  if (i >= NN * 32) return;
  int node = i >> 5, c = i & 31;
  float dv = dinv[node];
  out[i] = bias[c] + dv * dv * h[i];
}
__global__ void Argus_gcn_scatter(const int* __restrict__ ei, const float* __restrict__ h,
                                  const float* __restrict__ dinv, float* __restrict__ out) {
  int lane = threadIdx.x & 31;
  int e = blockIdx.x * 8 + (threadIdx.x >> 5);   // one wave per edge
  if (e >= NE) return;
  int src = ei[e], dst = ei[NE + e];
  float cf = dinv[src] * dinv[dst];
  atomicAdd(&out[(size_t)dst * 32 + lane], cf * h[(size_t)src * 32 + lane]);
}

// ---------------- edge MLP stage 1: r = relu(ea @ nnW1 + nnb1)  (E x 8) ----------------
__global__ void Argus_edge_r(const float* __restrict__ ea, const float* __restrict__ nnW1,
                             const float* __restrict__ nnb1, float* __restrict__ rbuf) {
  int e = blockIdx.x * blockDim.x + threadIdx.x;
  if (e >= NE) return;
  float a[16];
#pragma unroll
  for (int d = 0; d < 16; ++d) a[d] = ea[(size_t)e * 16 + d];
#pragma unroll
  for (int k = 0; k < 8; ++k) {
    float acc = nnb1[k];
#pragma unroll
    for (int d = 0; d < 16; ++d) acc += a[d] * nnW1[d * 8 + k];
    rbuf[(size_t)e * 8 + k] = fmaxf(acc, 0.0f);
  }
}

// ---------------- NNConv as GEMM: msg(E x 32) = [r (x) z | z](E x 288) @ [nnW2;nnb2] ----------------
__global__ void Argus_nnconv(const int* __restrict__ ei, const float* __restrict__ rbuf,
                             const float* __restrict__ z, const _Float16* __restrict__ frag,
                             float* __restrict__ agg, float* __restrict__ cnt) {
  __shared__ float s_z[8][16][32];
  __shared__ float s_r[8][16][8];
  __shared__ int   s_dst[8][16];
  int lane = threadIdx.x & 31;
  int wv   = threadIdx.x >> 5;
  int e0   = (blockIdx.x * 8 + wv) * 16;
  if (e0 >= NE) return;
  int m = lane & 15, half = lane >> 4;
  int e = e0 + m;
  int src = ei[e];
#pragma unroll
  for (int c = 0; c < 16; ++c)
    s_z[wv][m][half * 16 + c] = z[(size_t)src * 32 + half * 16 + c];
  if (half == 0) {
    int dst = ei[NE + e];
#pragma unroll
    for (int q = 0; q < 8; ++q) s_r[wv][m][q] = rbuf[(size_t)e * 8 + q];
    s_dst[wv][m] = dst;
    atomicAdd(&cnt[dst], 1.0f);
  }
  asm volatile("s_wait_dscnt 0x0" ::: "memory");   // LDS writes visible cross-lane (same wave)
  int kbase = (lane < 16) ? 0 : 8;
  float zv[16];
#pragma unroll
  for (int i = 0; i < 16; ++i) zv[i] = s_z[wv][m][kbase + ((i < 8) ? i : i + 8)];
  const v16h* Bf = (const v16h*)frag;
  v8f acc0 = {}, acc1 = {};
  for (int s = 0; s < 9; ++s) {             // K = 8*32 (r⊗z) + 32 (bias term, A=z)
    float rm = (s < 8) ? s_r[wv][m][s] : 1.0f;
    v16h a;
#pragma unroll
    for (int i = 0; i < 16; ++i) a[i] = (_Float16)(rm * zv[i]);
    v16h b0 = Bf[(s * 2 + 0) * 32 + lane];
    v16h b1 = Bf[(s * 2 + 1) * 32 + lane];
    acc0 = __builtin_amdgcn_wmma_f32_16x16x32_f16(false, a, false, b0, (short)0, acc0, false, false);
    acc1 = __builtin_amdgcn_wmma_f32_16x16x32_f16(false, a, false, b1, (short)0, acc1, false, false);
  }
  int n = lane & 15, mb = half * 8;
#pragma unroll
  for (int v = 0; v < 8; ++v) {
    int d = s_dst[wv][mb + v];
    atomicAdd(&agg[(size_t)d * 32 + n],      acc0[v]);
    atomicAdd(&agg[(size_t)d * 32 + 16 + n], acc1[v]);
  }
}

__global__ void Argus_nn_final(float* __restrict__ zs, const float* __restrict__ agg,
                               const float* __restrict__ cnt, const float* __restrict__ rootT,
                               const float* __restrict__ cbias, int t) {
  int i = blockIdx.x * blockDim.x + threadIdx.x;
  if (i >= NN * 32) return;
  int node = i >> 5, c = i & 31;
  float cc = fmaxf(cnt[node], 1.0f);
  zs[(size_t)node * 64 + t * 32 + c] = tanhf(agg[i] / cc + rootT[i] + cbias[c]);
}

// ---------------- sequential GRU scan over nodes (carry = 2x32) ----------------
__global__ void Argus_gru(const float* __restrict__ zs, const float* __restrict__ Wih,
                          const float* __restrict__ Whh, const float* __restrict__ bih,
                          const float* __restrict__ bhh, float* __restrict__ ys) {
  __shared__ float sx[64], sh[64], sgi[192], sgh[192];
  int tid = threadIdx.x;                 // blockDim = 192
  int t = tid / 96, j = tid % 96;
  float wih[32], whh[32];
#pragma unroll
  for (int k = 0; k < 32; ++k) { wih[k] = Wih[j * 32 + k]; whh[k] = Whh[j * 32 + k]; }
  float bi = bih[j], bh = bhh[j];
  if (tid < 64) sh[tid] = 0.0f;
  __syncthreads();
  for (int n = 0; n < NN; ++n) {
    if (tid < 64) sx[tid] = zs[(size_t)n * 64 + tid];
    __syncthreads();
    float a1 = bi, a2 = bh;
#pragma unroll
    for (int k = 0; k < 32; ++k) {
      a1 += sx[t * 32 + k] * wih[k];
      a2 += sh[t * 32 + k] * whh[k];
    }
    sgi[tid] = a1; sgh[tid] = a2;
    __syncthreads();
    if (tid < 64) {
      int tt = tid >> 5, c = tid & 31;
      float ir = sgi[tt * 96 + c],      hr = sgh[tt * 96 + c];
      float iz = sgi[tt * 96 + 32 + c], hz = sgh[tt * 96 + 32 + c];
      float in = sgi[tt * 96 + 64 + c], hn = sgh[tt * 96 + 64 + c];
      float rg = 1.0f / (1.0f + expf(-(ir + hr)));
      float ug = 1.0f / (1.0f + expf(-(iz + hz)));
      float ng = tanhf(in + rg * hn);
      float hv = (1.0f - ug) * ng + ug * sh[tid];
      sh[tid] = hv;
      ys[(size_t)n * 64 + tid] = hv;
    }
    __syncthreads();
  }
}

// ---------------- sampling + decode + softmax (wave per (node,t)) ----------------
__global__ void Argus_decode(const float* __restrict__ outlin, const int* __restrict__ idx,
                             const int* __restrict__ ptr, const float* __restrict__ Wdec,
                             const float* __restrict__ bdec, const float* __restrict__ blin,
                             float* __restrict__ dout,
                             unsigned fk00, unsigned fk01, unsigned fk10, unsigned fk11) {
  __shared__ float sW[1024], sbd[32], sbl[32];
  for (int i = threadIdx.x; i < 1024; i += blockDim.x) sW[i] = Wdec[i];
  if (threadIdx.x < 32) { sbd[threadIdx.x] = bdec[threadIdx.x]; sbl[threadIdx.x] = blin[threadIdx.x]; }
  __syncthreads();
  int lane = threadIdx.x & 31;
  int gid = blockIdx.x * 8 + (threadIdx.x >> 5);
  if (gid >= 2 * NN) return;
  int n = gid >> 1, t = gid & 1;
  unsigned k0 = t ? fk10 : fk00, k1 = t ? fk11 : fk01;
  int i0 = idx[n];
  float deg = (float)(idx[n + 1] - i0);
  float self = outlin[(size_t)n * 64 + t * 32 + lane];
  int nb = 0;
  if (lane < SS) {
    unsigned si = (unsigned)(n * SS + lane);          // (N,S) row-major
    const unsigned halfn = (NN * SS) / 2;
    unsigned p = (si < halfn) ? si : si - halfn;
    unsigned o0, o1; threefry2x32(k0, k1, p, p + halfn, &o0, &o1);
    unsigned bits = (si < halfn) ? o0 : o1;
    float u = __uint_as_float((bits >> 9) | 0x3f800000u) - 1.0f;
    int jj = (int)floorf(u * deg);
    int pos = i0 + jj;
    if (pos < 0) pos = 0;
    if (pos > NE - 1) pos = NE - 1;
    nb = ptr[pos];
  }
  float acc = self;
#pragma unroll
  for (int s = 0; s < SS; ++s) {
    int nbs = __shfl(nb, s, 32);
    acc += outlin[(size_t)nbs * 64 + t * 32 + lane];
  }
  float ag = ((deg > 0.0f) ? acc * (1.0f / 6.0f) : self) + sbl[lane];
  float d = sbd[lane];
#pragma unroll
  for (int k = 0; k < 32; ++k) d += __shfl(ag, k, 32) * sW[k * 32 + lane];
  float mx = d;
  for (int off = 16; off > 0; off >>= 1) mx = fmaxf(mx, __shfl_xor(mx, off, 32));
  float ex = expf(d - mx);
  float sm = ex;
  for (int off = 16; off > 0; off >>= 1) sm += __shfl_xor(sm, off, 32);
  dout[(size_t)t * NN * 32 + (size_t)n * 32 + lane] = ex / sm;
}

// ---------------- host orchestration ----------------
extern "C" void kernel_launch(void* const* d_in, const int* in_sizes, int n_in,
                              void* d_out, int out_size, void* d_ws, size_t ws_size,
                              hipStream_t stream) {
  (void)in_sizes; (void)n_in; (void)out_size; (void)ws_size;
  const float* x    = (const float*)d_in[0];
  const int*   ei0  = (const int*)  d_in[1];
  const float* ea0  = (const float*)d_in[2];
  const int*   ei1  = (const int*)  d_in[3];
  const float* ea1  = (const float*)d_in[4];
  const int*   idx  = (const int*)  d_in[5];
  const int*   ptr  = (const int*)  d_in[6];
  const float* W1   = (const float*)d_in[7];
  const float* b1   = (const float*)d_in[8];
  const float* W2   = (const float*)d_in[9];
  const float* b2   = (const float*)d_in[10];
  const float* W3   = (const float*)d_in[11];
  const float* b3   = (const float*)d_in[12];
  const float* nnW1 = (const float*)d_in[13];
  const float* nnb1 = (const float*)d_in[14];
  const float* nnW2 = (const float*)d_in[15];
  const float* nnb2 = (const float*)d_in[16];
  const float* root = (const float*)d_in[17];
  const float* cbias= (const float*)d_in[18];
  const float* Wih  = (const float*)d_in[19];
  const float* Whh  = (const float*)d_in[20];
  const float* bih  = (const float*)d_in[21];
  const float* bhh  = (const float*)d_in[22];
  const float* Wlin = (const float*)d_in[23];
  const float* blin = (const float*)d_in[24];
  const float* Wdec = (const float*)d_in[25];
  const float* bdec = (const float*)d_in[26];
  float* out = (float*)d_out;

  char* ws = (char*)d_ws; size_t off = 0;
  auto alloc = [&](size_t bytes) -> void* {
    void* p = ws + off; off += (bytes + 255) & ~(size_t)255; return p;
  };
  float* dinv   = (float*)alloc((size_t)NN * 4);
  float* hbuf   = (float*)alloc((size_t)NN * 32 * 4);
  float* zA     = (float*)alloc((size_t)NN * 32 * 4);
  float* zB     = (float*)alloc((size_t)NN * 32 * 4);
  float* rbuf   = (float*)alloc((size_t)NE * 8 * 4);
  float* agg    = (float*)alloc((size_t)NN * 32 * 4);
  float* cntdeg = (float*)alloc((size_t)NN * 4);
  float* rootT  = (float*)alloc((size_t)NN * 32 * 4);
  float* zs     = (float*)alloc((size_t)NN * 64 * 4);
  float* ysb    = (float*)alloc((size_t)NN * 64 * 4);
  float* outlin = (float*)alloc((size_t)NN * 64 * 4);
  _Float16* fW1   = (_Float16*)alloc(2 * 64 * 16 * 2);
  _Float16* fW2   = (_Float16*)alloc(64 * 16 * 2);
  _Float16* fW3   = (_Float16*)alloc(64 * 16 * 2);
  _Float16* fRoot = (_Float16*)alloc(64 * 16 * 2);
  _Float16* fLin  = (_Float16*)alloc(64 * 16 * 2);
  _Float16* fNN   = (_Float16*)alloc(9 * 64 * 16 * 2);

  Argus_packB<<<2, 64, 0, stream>>>(W1, fW1, 2);
  Argus_packB<<<1, 64, 0, stream>>>(W2, fW2, 1);
  Argus_packB<<<1, 64, 0, stream>>>(W3, fW3, 1);
  Argus_packB<<<1, 64, 0, stream>>>(root, fRoot, 1);
  Argus_packB<<<1, 64, 0, stream>>>(Wlin, fLin, 1);
  Argus_packNN<<<9, 64, 0, stream>>>(nnW2, nnb2, fNN);

  const int* eis[2] = { ei0, ei1 };
  const float* eas[2] = { ea0, ea1 };
  const float* gb[3] = { b1, b2, b3 };

  for (int t = 0; t < 2; ++t) {
    const int* ei = eis[t]; const float* ea = eas[t];
    Argus_fill<<<(NN + 255) / 256, 256, 0, stream>>>(cntdeg, 1.0f, NN);   // self loops
    Argus_count_deg<<<(NE + 255) / 256, 256, 0, stream>>>(ei, cntdeg);
    Argus_dinv<<<(NN + 255) / 256, 256, 0, stream>>>(cntdeg, dinv);

    const float* src = x; int ksteps = 2;
    float* dsts[3] = { zA, zB, zA };
    for (int layer = 0; layer < 3; ++layer) {
      Argus_gemm16<<<(NN / 16 + 7) / 8, 256, 0, stream>>>(src, layer == 0 ? fW1 : (layer == 1 ? fW2 : fW3),
                                                          hbuf, NN, ksteps);
      Argus_gcn_init<<<NN * 32 / 256, 256, 0, stream>>>(dsts[layer], hbuf, dinv, gb[layer]);
      Argus_gcn_scatter<<<NE / 8, 256, 0, stream>>>(ei, hbuf, dinv, dsts[layer]);
      if (layer > 0) Argus_relu<<<NN * 32 / 256, 256, 0, stream>>>(dsts[layer], NN * 32);
      src = dsts[layer]; ksteps = 1;
    }
    // zA now holds z3 = relu(gcn3)
    Argus_edge_r<<<(NE + 255) / 256, 256, 0, stream>>>(ea, nnW1, nnb1, rbuf);
    Argus_fill<<<NN * 32 / 256, 256, 0, stream>>>(agg, 0.0f, NN * 32);
    Argus_fill<<<(NN + 255) / 256, 256, 0, stream>>>(cntdeg, 0.0f, NN);
    Argus_nnconv<<<(NE / 16 + 7) / 8, 256, 0, stream>>>(ei, rbuf, zA, fNN, agg, cntdeg);
    Argus_gemm16<<<(NN / 16 + 7) / 8, 256, 0, stream>>>(zA, fRoot, rootT, NN, 1);
    Argus_nn_final<<<NN * 32 / 256, 256, 0, stream>>>(zs, agg, cntdeg, rootT, cbias, t);
  }

  Argus_gru<<<1, 192, 0, stream>>>(zs, Wih, Whh, bih, bhh, ysb);
  Argus_gemm16<<<(2 * NN / 16 + 7) / 8, 256, 0, stream>>>(ysb, fLin, outlin, 2 * NN, 1);

  unsigned fk[2][2];
  for (int t = 0; t < 2; ++t) threefry2x32(0u, 42u, 0u, (unsigned)t, &fk[t][0], &fk[t][1]);
  Argus_decode<<<(2 * NN / 8), 256, 0, stream>>>(outlin, idx, ptr, Wdec, bdec, blin, out,
                                                 fk[0][0], fk[0][1], fk[1][0], fk[1][1]);
}
